// SwinTransformerBlock3D_9964324126908
// MI455X (gfx1250) — matrix operand order repaired and checked by reading, not compiled, not run on hardware
//
#include <hip/hip_runtime.h>
#include <cmath>

typedef __attribute__((ext_vector_type(16))) __bf16        v16bf;
typedef __attribute__((ext_vector_type(8)))  float         v8f;
typedef __attribute__((ext_vector_type(4)))  unsigned int  v4u;

union FragU { v4u q[2]; v16bf v; __bf16 h[16]; };

#define BATCH 4
#define LTOK  10976
#define CH    96
#define HEADS 6
#define NTOK  343
#define NPAD  352
#define NTIL  22
#define BWIN  128
#define HID   384
#define MROWS (BWIN*NPAD)    /* 45056 */
#define TROWS (BATCH*LTOK)   /* 43904 */

__device__ __forceinline__ float wred(float v){
  #pragma unroll
  for (int o = 16; o > 0; o >>= 1) v += __shfl_xor(v, o, 32);
  return v;
}
__device__ __forceinline__ v8f wmma_bf16(v16bf a, v16bf b, v8f c){
  return __builtin_amdgcn_wmma_f32_16x16x32_bf16(false, a, false, b, (short)0, c, false, false);
}
// A-fragment (16x32 bf16, row-major source, 16B aligned): ISA layout reduces to
// two 128-bit loads: dword blocks {half, 2+half} of the 32-wide K chunk.
__device__ __forceinline__ v16bf load_a32(const __bf16* rp, int half){
  FragU f;
  const v4u* p = (const v4u*)rp;
  f.q[0] = p[half];
  f.q[1] = p[2 + half];
  return f.v;
}
// B-fragment from contiguous bf16 (same K order as A: dword blocks {half, 2+half}): two b128 loads.
__device__ __forceinline__ v16bf load_b32(const __bf16* bp, int half){
  FragU f;
  const v4u* p = (const v4u*)bp;
  f.q[0] = p[half];
  f.q[1] = p[2 + half];
  return f.v;
}
__device__ __forceinline__ int regD(int c){ return c < 7 ? 0 : (c < 11 ? 1 : 2); }
__device__ __forceinline__ int regHW(int c){ return c < 21 ? 0 : (c < 25 ? 1 : 2); }

// ---------------- one-time f32 -> bf16 weight conversion ----------------
__global__ __launch_bounds__(256) void wcvt_kernel(const float* __restrict__ s,
    __bf16* __restrict__ d, int n){
  int i = blockIdx.x * 256 + threadIdx.x;
  if (i < n) d[i] = (__bf16)s[i];
}

// ---------------- LN1 + shift-roll + window partition -> bf16 ----------------
__global__ __launch_bounds__(256) void ln1_part_kernel(const float* __restrict__ x,
    const float* __restrict__ g, const float* __restrict__ bta,
    __bf16* __restrict__ xw){
  int wid = threadIdx.x >> 5, lane = threadIdx.x & 31;
  int r = blockIdx.x * 8 + wid;
  if (r >= MROWS) return;
  __bf16* op = xw + (size_t)r * CH;
  int b_ = r / NPAD, tok = r % NPAD;
  if (tok >= NTOK){                          // zero the N padding rows
    for (int c = lane; c < CH; c += 32) op[c] = (__bf16)0.f;
    return;
  }
  int bb = b_ >> 5, wi = b_ & 31;
  int td = tok / 49, th = (tok / 7) % 7, tw = tok % 7;
  int gd = (wi >> 4) * 7 + td, gh = ((wi >> 2) & 3) * 7 + th, gw = (wi & 3) * 7 + tw;
  int sd = (gd + 3) % 14, sh = (gh + 3) % 28, sw = (gw + 3) % 28;   // roll -shift == gather at +shift
  const float* xp = x + ((size_t)bb * LTOK + (size_t)((sd * 28 + sh) * 28 + sw)) * CH;
  float v0 = xp[lane], v1 = xp[lane + 32], v2 = xp[lane + 64];
  float mu = wred(v0 + v1 + v2) * (1.f / 96.f);
  float d0 = v0 - mu, d1 = v1 - mu, d2 = v2 - mu;
  float var = wred(d0 * d0 + d1 * d1 + d2 * d2) * (1.f / 96.f);
  float inv = rsqrtf(var + 1e-5f);
  op[lane]      = (__bf16)(d0 * inv * g[lane]      + bta[lane]);
  op[lane + 32] = (__bf16)(d1 * inv * g[lane + 32] + bta[lane + 32]);
  op[lane + 64] = (__bf16)(d2 * inv * g[lane + 64] + bta[lane + 64]);
}

// ---------------- QKV GEMM: 3 N-tiles per wave, scatter per head ----------------
// q,k stored [which][b_][h][tok][16]; v stored transposed [b_][h][d][NPAD]
__global__ __launch_bounds__(32) void qkv_kernel(const __bf16* __restrict__ xw,
    const __bf16* __restrict__ wb, const float* __restrict__ bias,
    __bf16* __restrict__ qkvs){
  int mt = blockIdx.x, nt0 = blockIdx.y * 3;
  int lane = threadIdx.x, half = lane >> 4, ln = lane & 15;
  v8f acc[3];
  #pragma unroll
  for (int t = 0; t < 3; t++) acc[t] = (v8f){0.f,0.f,0.f,0.f,0.f,0.f,0.f,0.f};
  const __bf16* rp = xw + (size_t)(mt * 16 + ln) * CH;
  #pragma unroll
  for (int k0 = 0; k0 < CH; k0 += 32){
    v16bf a = load_a32(rp + k0, half);              // A reused across 3 N-tiles
    #pragma unroll
    for (int t = 0; t < 3; t++){
      const __bf16* wp = wb + (size_t)((nt0 + t) * 16 + ln) * CH + k0;
      acc[t] = wmma_bf16(a, load_b32(wp, half), acc[t]);
    }
  }
  #pragma unroll
  for (int t = 0; t < 3; t++){
    int oc = (nt0 + t) * 16 + ln;
    int which = oc / CH, hh = (oc % CH) / 16, dd = oc % 16;
    float bi = bias[oc];
    float sc = (which == 0) ? 0.25f : 1.0f;         // q * head_dim^-0.5
    #pragma unroll
    for (int r = 0; r < 8; r++){
      int m = mt * 16 + r + half * 8;
      int b_ = m / NPAD, tok = m % NPAD;
      float v = (acc[t][r] + bi) * sc;
      size_t idx;
      if (which == 2)  // v transposed: [2*BWIN..][b_][h][d][NPAD]
        idx = ((size_t)2 * BWIN * HEADS * NPAD * 16) +
              ((((size_t)b_ * HEADS + hh) * 16 + dd) * NPAD + tok);
      else
        idx = ((((size_t)which * BWIN + b_) * HEADS + hh) * NPAD + tok) * 16 + dd;
      qkvs[idx] = (__bf16)v;
    }
  }
}

// ---------------- fused attention: scores + bias + mask + softmax + PV ----------------
__global__ __launch_bounds__(32) void attn_kernel(const __bf16* __restrict__ qkvs,
    const float* __restrict__ rpb, __bf16* __restrict__ oat){
  __shared__ __align__(16) float  sbuf[16 * NPAD];
  __shared__ __align__(16) __bf16 pbuf[16 * NPAD];
  __shared__ int mdec[NPAD];   // packed md|mh<<4|mw<<8|region<<12 per key token
  __shared__ int qdec[16];
  int qt = blockIdx.x, h = blockIdx.y, b_ = blockIdx.z;
  int lane = threadIdx.x, half = lane >> 4, ln = lane & 15;
  int wi = b_ & 31, wd = wi >> 4, wh = (wi >> 2) & 3, ww = wi & 3;
  const __bf16* qb = qkvs + (((size_t)b_ * HEADS + h) * NPAD) * 16;
  const __bf16* kb = qb + (size_t)BWIN * HEADS * NPAD * 16;
  const __bf16* vt = qkvs + (size_t)2 * BWIN * HEADS * NPAD * 16 +
                     (((size_t)b_ * HEADS + h) * 16 + ln) * NPAD;   // row d=ln
  // decode tables
  for (int m = lane; m < NPAD; m += 32){
    if (m < NTOK){
      int md = m / 49, mh = (m / 7) % 7, mw = m % 7;
      int cm = regD(wd * 7 + md) * 9 + regHW(wh * 7 + mh) * 3 + regHW(ww * 7 + mw);
      mdec[m] = md | (mh << 4) | (mw << 8) | (cm << 12);
    } else mdec[m] = -1;
  }
  if (lane < 16){
    int qtok = qt * 16 + lane;
    int td = qtok / 49, th = (qtok / 7) % 7, tw = qtok % 7;
    int cq = regD(wd * 7 + td) * 9 + regHW(wh * 7 + th) * 3 + regHW(ww * 7 + tw);
    qdec[lane] = (qtok < NTOK) ? (td | (th << 4) | (tw << 8) | (cq << 12)) : -1;
  }
  // q fragment (head_dim 16 zero-padded to K=32): one b128 + zero
  FragU qf;
  {
    const v4u* p = (const v4u*)(qb + (size_t)(qt * 16 + ln) * 16);
    v4u z = {0u,0u,0u,0u};
    qf.q[0] = p[half];
    qf.q[1] = z;
  }
  // s = q @ k^T, 22 key tiles
  for (int kt = 0; kt < NTIL; kt++){
    FragU bf;
    v4u z = {0u,0u,0u,0u};
    {
      const v4u* p = (const v4u*)(kb + (size_t)(kt * 16 + ln) * 16);
      bf.q[0] = p[half]; bf.q[1] = z;
    }
    v8f c = {0.f,0.f,0.f,0.f,0.f,0.f,0.f,0.f};
    c = wmma_bf16(qf.v, bf.v, c);
    #pragma unroll
    for (int r = 0; r < 8; r++) sbuf[(r + half * 8) * NPAD + kt * 16 + ln] = c[r];
  }
  __syncthreads();
  // relative-position bias + shifted-window mask via decode tables
  for (int i = lane; i < 16 * NPAD; i += 32){
    int qi = i / NPAD, m = i % NPAD;
    int dm = mdec[m];
    if (dm < 0){ sbuf[i] = -1e30f; continue; }
    int dq = qdec[qi];
    if (dq < 0) continue;
    int rpi = ((dq & 15) - (dm & 15) + 6) * 169 +
              (((dq >> 4) & 15) - ((dm >> 4) & 15) + 6) * 13 +
              (((dq >> 8) & 15) - ((dm >> 8) & 15) + 6);
    float v = sbuf[i] + rpb[rpi * HEADS + h];
    if ((dq >> 12) != (dm >> 12)) v -= 100.f;
    sbuf[i] = v;
  }
  __syncthreads();
  // softmax per query row -> bf16 probabilities
  if (lane < 16){
    int base = lane * NPAD;
    float mx = -1e30f;
    for (int m = 0; m < NPAD; m++) mx = fmaxf(mx, sbuf[base + m]);
    float s = 0.f;
    for (int m = 0; m < NPAD; m++){ float e = __expf(sbuf[base + m] - mx); sbuf[base + m] = e; s += e; }
    float inv = 1.f / s;
    for (int m = 0; m < NPAD; m++) pbuf[base + m] = (__bf16)(sbuf[base + m] * inv);
  }
  __syncthreads();
  // o = p @ v, K = 352 = 11 x 32; P from LDS (ds_load_b128), V^T rows contiguous
  v8f acc = {0.f,0.f,0.f,0.f,0.f,0.f,0.f,0.f};
  const __bf16* pr = &pbuf[ln * NPAD];
  #pragma unroll
  for (int mk = 0; mk < NPAD / 32; mk++)
    acc = wmma_bf16(load_a32(pr + mk * 32, half), load_b32(vt + mk * 32, half), acc);
  #pragma unroll
  for (int r = 0; r < 8; r++){
    int tok = qt * 16 + r + half * 8;
    oat[((size_t)b_ * NPAD + tok) * CH + h * 16 + ln] = (__bf16)acc[r];
  }
}

// ---------------- proj GEMM (3 N-tiles/wave) + window-reverse + residual ----------------
__global__ __launch_bounds__(32) void proj_kernel(const __bf16* __restrict__ oat,
    const __bf16* __restrict__ wb, const float* __restrict__ bias,
    const float* __restrict__ x, float* __restrict__ y){
  int mt = blockIdx.x, nt0 = blockIdx.y * 3;
  int lane = threadIdx.x, half = lane >> 4, ln = lane & 15;
  v8f acc[3];
  #pragma unroll
  for (int t = 0; t < 3; t++) acc[t] = (v8f){0.f,0.f,0.f,0.f,0.f,0.f,0.f,0.f};
  const __bf16* rp = oat + (size_t)(mt * 16 + ln) * CH;
  #pragma unroll
  for (int k0 = 0; k0 < CH; k0 += 32){
    v16bf a = load_a32(rp + k0, half);
    #pragma unroll
    for (int t = 0; t < 3; t++){
      const __bf16* wp = wb + (size_t)((nt0 + t) * 16 + ln) * CH + k0;
      acc[t] = wmma_bf16(a, load_b32(wp, half), acc[t]);
    }
  }
  #pragma unroll
  for (int r = 0; r < 8; r++){
    int m = mt * 16 + r + half * 8;
    int b_ = m / NPAD, tok = m % NPAD;
    if (tok >= NTOK) continue;
    int bb = b_ >> 5, wi = b_ & 31;
    int td = tok / 49, th = (tok / 7) % 7, tw = tok % 7;
    int gd = (wi >> 4) * 7 + td, gh = ((wi >> 2) & 3) * 7 + th, gw = (wi & 3) * 7 + tw;
    int sd = (gd + 3) % 14, sh = (gh + 3) % 28, sw = (gw + 3) % 28;  // roll back by +shift
    size_t base = ((size_t)bb * LTOK + (size_t)((sd * 28 + sh) * 28 + sw)) * CH;
    #pragma unroll
    for (int t = 0; t < 3; t++){
      int c = (nt0 + t) * 16 + ln;
      y[base + c] = x[base + c] + acc[t][r] + bias[c];
    }
  }
}

// ---------------- LN2 -> bf16 ----------------
__global__ __launch_bounds__(256) void ln2_kernel(const float* __restrict__ y,
    const float* __restrict__ g, const float* __restrict__ bta,
    __bf16* __restrict__ mbf){
  int wid = threadIdx.x >> 5, lane = threadIdx.x & 31;
  int r = blockIdx.x * 8 + wid;
  if (r >= TROWS) return;
  const float* yp = y + (size_t)r * CH;
  float v0 = yp[lane], v1 = yp[lane + 32], v2 = yp[lane + 64];
  float mu = wred(v0 + v1 + v2) * (1.f / 96.f);
  float d0 = v0 - mu, d1 = v1 - mu, d2 = v2 - mu;
  float var = wred(d0 * d0 + d1 * d1 + d2 * d2) * (1.f / 96.f);
  float inv = rsqrtf(var + 1e-5f);
  __bf16* op = mbf + (size_t)r * CH;
  op[lane]      = (__bf16)(d0 * inv * g[lane]      + bta[lane]);
  op[lane + 32] = (__bf16)(d1 * inv * g[lane + 32] + bta[lane + 32]);
  op[lane + 64] = (__bf16)(d2 * inv * g[lane + 64] + bta[lane + 64]);
}

// ---------------- fc1 GEMM (4 N-tiles/wave) + exact GELU ----------------
__global__ __launch_bounds__(32) void fc1_kernel(const __bf16* __restrict__ mbf,
    const __bf16* __restrict__ wb, const float* __restrict__ bias,
    __bf16* __restrict__ h1){
  int mt = blockIdx.x, nt0 = blockIdx.y * 4;
  int lane = threadIdx.x, half = lane >> 4, ln = lane & 15;
  v8f acc[4];
  #pragma unroll
  for (int t = 0; t < 4; t++) acc[t] = (v8f){0.f,0.f,0.f,0.f,0.f,0.f,0.f,0.f};
  const __bf16* rp = mbf + (size_t)(mt * 16 + ln) * CH;
  #pragma unroll
  for (int k0 = 0; k0 < CH; k0 += 32){
    v16bf a = load_a32(rp + k0, half);
    #pragma unroll
    for (int t = 0; t < 4; t++){
      const __bf16* wp = wb + (size_t)((nt0 + t) * 16 + ln) * CH + k0;
      acc[t] = wmma_bf16(a, load_b32(wp, half), acc[t]);
    }
  }
  #pragma unroll
  for (int t = 0; t < 4; t++){
    int c = (nt0 + t) * 16 + ln;
    float bi = bias[c];
    #pragma unroll
    for (int r = 0; r < 8; r++){
      int m = mt * 16 + r + half * 8;
      float v = acc[t][r] + bi;
      v = 0.5f * v * (1.f + erff(v * 0.70710678118654752f));   // exact GELU
      h1[(size_t)m * HID + c] = (__bf16)v;
    }
  }
}

// ---------------- fc2 GEMM (3 N-tiles/wave) + residual -> out ----------------
__global__ __launch_bounds__(32) void fc2_kernel(const __bf16* __restrict__ h1,
    const __bf16* __restrict__ wb, const float* __restrict__ bias,
    const float* __restrict__ y, float* __restrict__ out){
  int mt = blockIdx.x, nt0 = blockIdx.y * 3;
  int lane = threadIdx.x, half = lane >> 4, ln = lane & 15;
  v8f acc[3];
  #pragma unroll
  for (int t = 0; t < 3; t++) acc[t] = (v8f){0.f,0.f,0.f,0.f,0.f,0.f,0.f,0.f};
  const __bf16* rp = h1 + (size_t)(mt * 16 + ln) * HID;
  #pragma unroll 4
  for (int k0 = 0; k0 < HID; k0 += 32){
    v16bf a = load_a32(rp + k0, half);
    #pragma unroll
    for (int t = 0; t < 3; t++){
      const __bf16* wp = wb + (size_t)((nt0 + t) * 16 + ln) * HID + k0;
      acc[t] = wmma_bf16(a, load_b32(wp, half), acc[t]);
    }
  }
  #pragma unroll
  for (int t = 0; t < 3; t++){
    int c = (nt0 + t) * 16 + ln;
    float bi = bias[c];
    #pragma unroll
    for (int r = 0; r < 8; r++){
      int m = mt * 16 + r + half * 8;
      size_t idx = (size_t)m * CH + c;
      out[idx] = y[idx] + acc[t][r] + bi;
    }
  }
}

extern "C" void kernel_launch(void* const* d_in, const int* in_sizes, int n_in,
                              void* d_out, int out_size, void* d_ws, size_t ws_size,
                              hipStream_t stream){
  const float* x      = (const float*)d_in[0];
  const float* n1g    = (const float*)d_in[4];
  const float* n1b    = (const float*)d_in[5];
  const float* qkv_w  = (const float*)d_in[6];
  const float* qkv_b  = (const float*)d_in[7];
  const float* rpb    = (const float*)d_in[8];
  const float* proj_w = (const float*)d_in[9];
  const float* proj_b = (const float*)d_in[10];
  const float* n2g    = (const float*)d_in[11];
  const float* n2b    = (const float*)d_in[12];
  const float* fc1_w  = (const float*)d_in[13];
  const float* fc1_b  = (const float*)d_in[14];
  const float* fc2_w  = (const float*)d_in[15];
  const float* fc2_b  = (const float*)d_in[16];
  float* out = (float*)d_out;

  // workspace layout (bytes)
  char* ws = (char*)d_ws;
  __bf16* xw    = (__bf16*)(ws);                //  8,650,752  [45056][96]
  __bf16* qkvs  = (__bf16*)(ws + 8650752);      // 25,952,256  q,k:[128][6][352][16]  vT:[128][6][16][352]
  __bf16* oat   = (__bf16*)(ws + 34603008);     //  8,650,752  [45056][96]
  float*  y     = (float*) (ws + 43253760);     // 16,859,136  [43904][96]
  __bf16* mbf   = (__bf16*)(ws + 60112896);     //  8,429,568  [43904][96]
  __bf16* h1    = (__bf16*)(ws + 68542464);     // 33,718,272  [43904][384]
  __bf16* qkvwb = (__bf16*)(ws + 102260736);    //     55,296  [288][96]
  __bf16* prjwb = (__bf16*)(ws + 102316032);    //     18,432  [96][96]
  __bf16* fc1wb = (__bf16*)(ws + 102334464);    //     73,728  [384][96]
  __bf16* fc2wb = (__bf16*)(ws + 102408192);    //     73,728  [96][384]

  // one-time bf16 weight packing (removes f32 loads + cvt from all GEMM loops)
  wcvt_kernel<<<dim3(108), 256, 0, stream>>>(qkv_w,  qkvwb, 27648);
  wcvt_kernel<<<dim3(36),  256, 0, stream>>>(proj_w, prjwb, 9216);
  wcvt_kernel<<<dim3(144), 256, 0, stream>>>(fc1_w,  fc1wb, 36864);
  wcvt_kernel<<<dim3(144), 256, 0, stream>>>(fc2_w,  fc2wb, 36864);

  ln1_part_kernel<<<dim3(MROWS / 8), 256, 0, stream>>>(x, n1g, n1b, xw);
  qkv_kernel<<<dim3(MROWS / 16, 6), 32, 0, stream>>>(xw, qkvwb, qkv_b, qkvs);
  attn_kernel<<<dim3(NTIL, HEADS, BWIN), 32, 0, stream>>>(qkvs, rpb, oat);
  proj_kernel<<<dim3(MROWS / 16, 2), 32, 0, stream>>>(oat, prjwb, proj_b, x, y);
  ln2_kernel<<<dim3(TROWS / 8), 256, 0, stream>>>(y, n2g, n2b, mbf);
  fc1_kernel<<<dim3(TROWS / 16, 6), 32, 0, stream>>>(mbf, fc1wb, fc1_b, h1);
  fc2_kernel<<<dim3(TROWS / 16, 2), 32, 0, stream>>>(h1, fc2wb, fc2_b, y, out);
}
